// DepthwiseXCorr_70394513981952
// MI455X (gfx1250) — matrix operand order repaired and checked by reading, not compiled, not run on hardware
//
#include <hip/hip_runtime.h>

typedef __attribute__((ext_vector_type(16))) _Float16 v16h;
typedef __attribute__((ext_vector_type(8)))  float    v8f;
typedef __attribute__((ext_vector_type(4)))  unsigned int u32x4;
typedef __attribute__((__vector_size__(16))) int i32x4v;   // matches builtin param type

// --- optional CDNA5 async global->LDS path (guarded; falls back to copies) ---
#ifndef USE_ASYNC_LDS
#define USE_ASYNC_LDS 1
#endif
#if USE_ASYNC_LDS && defined(__gfx1250__) && __has_builtin(__builtin_amdgcn_global_load_async_to_lds_b128)
#define HAVE_ASYNC 1
#else
#define HAVE_ASYNC 0
#endif

__device__ __forceinline__ void async_ldscpy_b128(const _Float16* g, _Float16* l) {
#if HAVE_ASYNC
    __builtin_amdgcn_global_load_async_to_lds_b128((i32x4v*)g, (i32x4v*)l, 0, 0);
#else
    (void)g; (void)l;
#endif
}

__device__ __forceinline__ void wait_async_lds() {
#if HAVE_ASYNC
#if __has_builtin(__builtin_amdgcn_s_wait_asynccnt)
    __builtin_amdgcn_s_wait_asynccnt(0);
#else
    asm volatile("s_wait_asynccnt 0x0" ::: "memory");
#endif
#endif
}

// ---------------------------------------------------------------------------
// WMMA fragment loaders (wave32, 16x16x32 f16 layouts per CDNA5 ISA 7.12.2)
// ---------------------------------------------------------------------------
__device__ __forceinline__ v16h load_frag_a(const _Float16* base, int lane) {
    int row = lane & 15;
    int hi  = (lane >> 4) & 1;
    const _Float16* p = base + row * 32 + hi * 8;
    union { v16h v; u32x4 u[2]; } r;
    r.u[0] = *(const u32x4*)(p);        // K pairs 0..7 (or 8..15)
    r.u[1] = *(const u32x4*)(p + 16);   // K pairs 16..23 (or 24..31)
    return r.v;
}

__device__ __forceinline__ v16h load_frag_b(const _Float16* colbase, int lane) {
    int n  = lane & 15;
    int hi = (lane >> 4) & 1;
    const _Float16* p = colbase + n * 32 + hi * 8;
    union { v16h v; u32x4 u[2]; } r;
    r.u[0] = *(const u32x4*)(p);
    r.u[1] = *(const u32x4*)(p + 16);
    return r.v;
}

// ---------------------------------------------------------------------------
// Prep kernels
// ---------------------------------------------------------------------------
__global__ void __launch_bounds__(256)
prep_w3x3(const float* __restrict__ w, _Float16* __restrict__ o) {
    // in [O=256][I=256][3][3] -> out [O][k], k = r*256 + ci (r = kh*3+kw)
    int idx = blockIdx.x * 256 + threadIdx.x;
    if (idx >= 256 * 2304) return;
    int oc = idx / 2304;
    int k  = idx % 2304;
    int r  = k / 256;
    int ci = k % 256;
    o[idx] = (_Float16)w[oc * 2304 + ci * 9 + r];
}

__global__ void __launch_bounds__(256)
prep_w1x1(const float* __restrict__ w, _Float16* __restrict__ o) {
    int idx = blockIdx.x * 256 + threadIdx.x;
    if (idx < 65536) o[idx] = (_Float16)w[idx];
}

__global__ void __launch_bounds__(256)
prep_bn(const float* __restrict__ g, const float* __restrict__ b,
        const float* __restrict__ m, const float* __restrict__ v,
        float* __restrict__ sc, float* __restrict__ sh) {
    int i = threadIdx.x;
    float s = g[i] * rsqrtf(v[i] + 1e-5f);
    sc[i] = s;
    sh[i] = b[i] - m[i] * s;
}

// ---------------------------------------------------------------------------
// Implicit-GEMM 3x3 conv + BN + ReLU, f16 WMMA, f32 accum, double-buffered LDS.
// Per WG (256 thr / 8 waves): 32 M (spatial) x 256 N (out-ch), one image.
// K = 2304 in 72 chunks of 32 (chunk c: r = c>>3, ci-block = c&7).
// ---------------------------------------------------------------------------
template<int Hin, int Win, int Hout, int Wout>
__global__ void __launch_bounds__(256)
conv3x3_bn_relu_wmma(const float* __restrict__ x, const _Float16* __restrict__ Bw,
                     const float* __restrict__ scale, const float* __restrict__ shift,
                     _Float16* __restrict__ out) {
    constexpr int S  = Hout * Wout;
    constexpr int NC = 72;
    const int b     = blockIdx.y;
    const int mBase = blockIdx.x * 32;
    const int tid   = threadIdx.x;
    const int lane  = tid & 31;
    const int wave  = tid >> 5;
    const int mt    = wave & 1;
    const int nw    = wave >> 1;

    // double buffer: buf = 1024 (A) + 8192 (B) halves; x2 = 36864 bytes
    __shared__ unsigned long long smem_u64[36864 / 8];
    _Float16* smem = (_Float16*)smem_u64;

    v8f acc[4] = {};
    const float* xb = x + (size_t)b * 256 * Hin * Win;

    auto loadChunk = [&](int c) {
        _Float16* aL = smem + (c & 1) * 9216;
        _Float16* bL = aL + 1024;
        const int r = c >> 3, cb = c & 7;
        const int kh = r / 3, kw = r % 3;
        // A: im2col gather, 1024 elems, fp32 -> f16
        #pragma unroll
        for (int i = 0; i < 4; ++i) {
            int e  = tid * 4 + i;
            int ml = e >> 5, kk = e & 31;
            int m  = mBase + ml;
            if (m >= S) m = 0;                  // masked at store time
            int oh = m / Wout, ow = m % Wout;
            int ci = cb * 32 + kk;
            float v = xb[((size_t)ci * Hin + (oh + kh)) * Win + (ow + kw)];
            aL[ml * 32 + kk] = (_Float16)v;
        }
        // B: 256 cols x 32 K, 64B per thread
        const _Float16* src = Bw + (size_t)tid * 2304 + r * 256 + cb * 32;
        _Float16* dst = bL + tid * 32;
#if HAVE_ASYNC
        async_ldscpy_b128(src,      dst);
        async_ldscpy_b128(src +  8, dst +  8);
        async_ldscpy_b128(src + 16, dst + 16);
        async_ldscpy_b128(src + 24, dst + 24);
#else
        const u32x4* s4 = (const u32x4*)src;
        u32x4* d4 = (u32x4*)dst;
        d4[0] = s4[0]; d4[1] = s4[1]; d4[2] = s4[2]; d4[3] = s4[3];
        if (c + 2 < NC) {
            const int r2 = (c + 2) >> 3, cb2 = (c + 2) & 7;
            __builtin_prefetch(Bw + (size_t)tid * 2304 + r2 * 256 + cb2 * 32, 0, 0);
        }
#endif
    };

    auto computeChunk = [&](int c) {
        _Float16* aL = smem + (c & 1) * 9216;
        _Float16* bL = aL + 1024;
        v16h af  = load_frag_a(aL + mt * 512, lane);
        v16h bf0 = load_frag_b(bL + (nw * 64 +  0) * 32, lane);
        v16h bf1 = load_frag_b(bL + (nw * 64 + 16) * 32, lane);
        v16h bf2 = load_frag_b(bL + (nw * 64 + 32) * 32, lane);
        v16h bf3 = load_frag_b(bL + (nw * 64 + 48) * 32, lane);
        acc[0] = __builtin_amdgcn_wmma_f32_16x16x32_f16(false, af, false, bf0, (short)0, acc[0], false, false);
        acc[1] = __builtin_amdgcn_wmma_f32_16x16x32_f16(false, af, false, bf1, (short)0, acc[1], false, false);
        acc[2] = __builtin_amdgcn_wmma_f32_16x16x32_f16(false, af, false, bf2, (short)0, acc[2], false, false);
        acc[3] = __builtin_amdgcn_wmma_f32_16x16x32_f16(false, af, false, bf3, (short)0, acc[3], false, false);
    };

    loadChunk(0);
    wait_async_lds();
    __syncthreads();
    for (int c = 0; c < NC; ++c) {
        if (c + 1 < NC) loadChunk(c + 1);
        computeChunk(c);
        wait_async_lds();
        __syncthreads();
    }

    // Epilogue: BN + ReLU -> padded LDS transpose -> coalesced NCHW f16 store
    {
        const int col = lane & 15, hi = lane >> 4;
        #pragma unroll
        for (int s = 0; s < 4; ++s) {
            int n = nw * 64 + s * 16 + col;
            float sc = scale[n], sh = shift[n];
            #pragma unroll
            for (int g = 0; g < 8; ++g) {
                float v = acc[s][g] * sc + sh;
                v = v > 0.f ? v : 0.f;
                int ml = mt * 16 + g + hi * 8;
                smem[n * 34 + ml] = (_Float16)v;
            }
        }
    }
    __syncthreads();
    for (int pass = 0; pass < 32; ++pass) {
        int ml = lane;
        int n  = wave + pass * 8;
        int m  = mBase + ml;
        if (m < S) out[((size_t)b * 256 + n) * S + m] = smem[n * 34 + ml];
    }
}

// ---------------------------------------------------------------------------
// Depthwise cross-correlation: s[29x29] (x) k[5x5] -> feat[25x25], f16 in/out
// ---------------------------------------------------------------------------
__global__ void __launch_bounds__(256)
xcorr_dw(const _Float16* __restrict__ s, const _Float16* __restrict__ k,
         _Float16* __restrict__ feat) {
    const int c = blockIdx.x, b = blockIdx.y;
    const size_t plane = (size_t)b * 256 + c;
    const _Float16* sp = s + plane * 841;
    const _Float16* kp = k + plane * 25;
    __shared__ _Float16 sl[841];
    __shared__ float    kl[25];
    const int tid = threadIdx.x;
    for (int i = tid; i < 841; i += 256) sl[i] = sp[i];
    if (tid < 25) kl[tid] = (float)kp[tid];
    __syncthreads();
    for (int o = tid; o < 625; o += 256) {
        int oh = o / 25, ow = o % 25;
        float acc = 0.f;
        #pragma unroll
        for (int i = 0; i < 5; ++i)
            #pragma unroll
            for (int j = 0; j < 5; ++j)
                acc += (float)sl[(oh + i) * 29 + ow + j] * kl[i * 5 + j];
        feat[plane * 625 + o] = (_Float16)acc;
    }
}

// ---------------------------------------------------------------------------
// Fused head: 1x1 conv (WMMA, K=256, double-buffered) + BN + ReLU -> h in LDS,
// then N=10 1x1 conv (+bias) as VALU reduction over LDS h. f32 output.
// ---------------------------------------------------------------------------
__global__ void __launch_bounds__(256)
head_wmma(const _Float16* __restrict__ feat, const _Float16* __restrict__ Bh1,
          const float* __restrict__ scale, const float* __restrict__ shift,
          const float* __restrict__ w2, const float* __restrict__ bias,
          float* __restrict__ outp) {
    constexpr int S  = 625;
    constexpr int NC = 8;
    const int b     = blockIdx.y;
    const int mBase = blockIdx.x * 32;
    const int tid   = threadIdx.x;
    const int lane  = tid & 31;
    const int wave  = tid >> 5;
    const int mt    = wave & 1;
    const int nw    = wave >> 1;

    __shared__ unsigned long long smem_u64[36864 / 8];
    _Float16* smem = (_Float16*)smem_u64;

    v8f acc[4] = {};
    const _Float16* fb = feat + (size_t)b * 256 * S;

    auto loadChunk = [&](int c) {
        _Float16* aL = smem + (c & 1) * 9216;
        _Float16* bL = aL + 1024;
        #pragma unroll
        for (int i = 0; i < 4; ++i) {
            int e  = tid * 4 + i;
            int ml = e >> 5, kk = e & 31;
            int m  = mBase + ml;
            if (m >= S) m = 0;
            int ci = c * 32 + kk;
            aL[ml * 32 + kk] = fb[(size_t)ci * S + m];
        }
        const _Float16* src = Bh1 + (size_t)tid * 256 + c * 32;
        _Float16* dst = bL + tid * 32;
#if HAVE_ASYNC
        async_ldscpy_b128(src,      dst);
        async_ldscpy_b128(src +  8, dst +  8);
        async_ldscpy_b128(src + 16, dst + 16);
        async_ldscpy_b128(src + 24, dst + 24);
#else
        const u32x4* s4 = (const u32x4*)src;
        u32x4* d4 = (u32x4*)dst;
        d4[0] = s4[0]; d4[1] = s4[1]; d4[2] = s4[2]; d4[3] = s4[3];
#endif
    };

    auto computeChunk = [&](int c) {
        _Float16* aL = smem + (c & 1) * 9216;
        _Float16* bL = aL + 1024;
        v16h af  = load_frag_a(aL + mt * 512, lane);
        v16h bf0 = load_frag_b(bL + (nw * 64 +  0) * 32, lane);
        v16h bf1 = load_frag_b(bL + (nw * 64 + 16) * 32, lane);
        v16h bf2 = load_frag_b(bL + (nw * 64 + 32) * 32, lane);
        v16h bf3 = load_frag_b(bL + (nw * 64 + 48) * 32, lane);
        acc[0] = __builtin_amdgcn_wmma_f32_16x16x32_f16(false, af, false, bf0, (short)0, acc[0], false, false);
        acc[1] = __builtin_amdgcn_wmma_f32_16x16x32_f16(false, af, false, bf1, (short)0, acc[1], false, false);
        acc[2] = __builtin_amdgcn_wmma_f32_16x16x32_f16(false, af, false, bf2, (short)0, acc[2], false, false);
        acc[3] = __builtin_amdgcn_wmma_f32_16x16x32_f16(false, af, false, bf3, (short)0, acc[3], false, false);
    };

    loadChunk(0);
    wait_async_lds();
    __syncthreads();
    for (int c = 0; c < NC; ++c) {
        if (c + 1 < NC) loadChunk(c + 1);
        computeChunk(c);
        wait_async_lds();
        __syncthreads();
    }

    // BN3 + ReLU -> h in LDS [channel n][ml] (stride 34 halves)
    {
        const int col = lane & 15, hi = lane >> 4;
        #pragma unroll
        for (int s = 0; s < 4; ++s) {
            int n = nw * 64 + s * 16 + col;
            float sc = scale[n], sh = shift[n];
            #pragma unroll
            for (int g = 0; g < 8; ++g) {
                float v = acc[s][g] * sc + sh;
                v = v > 0.f ? v : 0.f;
                int ml = mt * 16 + g + hi * 8;
                smem[n * 34 + ml] = (_Float16)v;
            }
        }
    }
    __syncthreads();
    // Second 1x1 conv: out[m][o] = sum_c h[c][m] * w2[o][c] + bias[o]
    for (int task = tid; task < 320; task += 256) {
        int o  = task >> 5;
        int ml = task & 31;
        int m  = mBase + ml;
        float sum = bias[o];
        const float* wo = w2 + o * 256;
        #pragma unroll 8
        for (int c = 0; c < 256; ++c)
            sum += (float)smem[c * 34 + ml] * wo[c];
        if (m < S) outp[((size_t)b * 10 + o) * S + m] = sum;
    }
}

// ---------------------------------------------------------------------------
extern "C" void kernel_launch(void* const* d_in, const int* in_sizes, int n_in,
                              void* d_out, int out_size, void* d_ws, size_t ws_size,
                              hipStream_t stream) {
    const float* x_k  = (const float*)d_in[0];   // [128,256,7,7]
    const float* x_s  = (const float*)d_in[1];   // [128,256,31,31]
    const float* w_ck = (const float*)d_in[2];
    const float* g1 = (const float*)d_in[3],  *b1 = (const float*)d_in[4];
    const float* m1 = (const float*)d_in[5],  *v1 = (const float*)d_in[6];
    const float* w_cs = (const float*)d_in[7];
    const float* g2 = (const float*)d_in[8],  *b2 = (const float*)d_in[9];
    const float* m2 = (const float*)d_in[10], *v2 = (const float*)d_in[11];
    const float* w_h1 = (const float*)d_in[12];
    const float* g3 = (const float*)d_in[13], *b3 = (const float*)d_in[14];
    const float* m3 = (const float*)d_in[15], *v3 = (const float*)d_in[16];
    const float* w_h2 = (const float*)d_in[17];
    const float* b_h2 = (const float*)d_in[18];
    float* out = (float*)d_out;                  // [128,10,25,25]

    char* ws = (char*)d_ws;
    _Float16* Bck  = (_Float16*)(ws + 0);              // 256*2304 halves
    _Float16* Bcs  = (_Float16*)(ws + 1179648);
    _Float16* Bh1  = (_Float16*)(ws + 2359296);        // 256*256 halves
    float*    sc1  = (float*)   (ws + 2490368);
    float*    sh1  = (float*)   (ws + 2491392);
    float*    sc2  = (float*)   (ws + 2492416);
    float*    sh2  = (float*)   (ws + 2493440);
    float*    sc3  = (float*)   (ws + 2494464);
    float*    sh3  = (float*)   (ws + 2495488);
    _Float16* kout = (_Float16*)(ws + 2496512);        // [128,256,25]
    _Float16* sout = (_Float16*)(ws + 4134912);        // [128,256,841]
    _Float16* feat = (_Float16*)(ws + 59250688);       // [128,256,625]

    prep_w3x3<<<2304, 256, 0, stream>>>(w_ck, Bck);
    prep_w3x3<<<2304, 256, 0, stream>>>(w_cs, Bcs);
    prep_w1x1<<<256, 256, 0, stream>>>(w_h1, Bh1);
    prep_bn<<<1, 256, 0, stream>>>(g1, b1, m1, v1, sc1, sh1);
    prep_bn<<<1, 256, 0, stream>>>(g2, b2, m2, v2, sc2, sh2);
    prep_bn<<<1, 256, 0, stream>>>(g3, b3, m3, v3, sc3, sh3);

    conv3x3_bn_relu_wmma<7, 7, 5, 5>
        <<<dim3(1, 128), 256, 0, stream>>>(x_k, Bck, sc1, sh1, kout);
    conv3x3_bn_relu_wmma<31, 31, 29, 29>
        <<<dim3(27, 128), 256, 0, stream>>>(x_s, Bcs, sc2, sh2, sout);

    xcorr_dw<<<dim3(256, 128), 256, 0, stream>>>(sout, kout, feat);

    head_wmma<<<dim3(20, 128), 256, 0, stream>>>(feat, Bh1, sc3, sh3, w_h2, b_h2, out);
}
